// mySGConv_86217173500064
// MI455X (gfx1250) — matrix operand order repaired and checked by reading, not compiled, not run on hardware
//
#include <hip/hip_runtime.h>
#include <hip/hip_bf16.h>

#define N_NODES 10000
#define N_EDGES 160000
#define C_DIM   512
#define OUT_COLS 2048
#define M_PAD   10048          // 157 * 64, GEMM row padding

typedef __attribute__((ext_vector_type(16))) __bf16 v16bf;
typedef __attribute__((ext_vector_type(8)))  __bf16 v8bf;
typedef __attribute__((ext_vector_type(4)))  __bf16 v4bf;
typedef __attribute__((ext_vector_type(8)))  float  v8f;
typedef __attribute__((ext_vector_type(4)))  int    v4i;

// async-to-LDS builtin wants int4* in explicit address spaces (AS1 global, AS3 LDS)
#define GAS(p) ((__attribute__((address_space(1))) v4i*)(p))
#define LAS(p) ((__attribute__((address_space(3))) v4i*)(p))

// ---------------------------------------------------------------------------
// Graph preprocessing: CSR over destination (col) + gcn_norm weights
// ---------------------------------------------------------------------------

__global__ void sg_init_kernel(int* __restrict__ counts, int* __restrict__ fill,
                               float* __restrict__ deg1, float* __restrict__ deg2) {
    int i = blockIdx.x * blockDim.x + threadIdx.x;
    if (i < N_NODES) {
        counts[i] = 0;
        fill[i]   = 0;
        deg1[i]   = 1.0f;   // self-loop weight 1
        deg2[i]   = 1.0f;
    }
}

__global__ void sg_edge_count_kernel(const int* __restrict__ ei, const float* __restrict__ ef,
                                     int* __restrict__ counts,
                                     float* __restrict__ deg1, float* __restrict__ deg2) {
    int e = blockIdx.x * blockDim.x + threadIdx.x;
    if (e < N_EDGES) {
        int c = ei[N_EDGES + e];
        atomicAdd(&counts[c], 1);
        atomicAdd(&deg1[c], fabsf(ef[e * 4 + 0]));
        atomicAdd(&deg2[c], fabsf(ef[e * 4 + 1]));
    }
}

__global__ void sg_dinv_kernel(float* __restrict__ deg1, float* __restrict__ deg2) {
    int i = blockIdx.x * blockDim.x + threadIdx.x;
    if (i < N_NODES) {
        float d1 = deg1[i];
        float d2 = deg2[i];
        deg1[i] = (d1 > 0.0f) ? rsqrtf(d1) : 0.0f;   // in-place deg -> dinv
        deg2[i] = (d2 > 0.0f) ? rsqrtf(d2) : 0.0f;
    }
}

// single-block exclusive scan of counts[N] -> col_ptr[N+1]
__global__ void sg_scan_kernel(const int* __restrict__ counts, int* __restrict__ col_ptr) {
    __shared__ int buf[1024];
    __shared__ int carry;
    int tid = threadIdx.x;
    if (tid == 0) carry = 0;
    __syncthreads();
    for (int base = 0; base < N_NODES; base += 1024) {
        int v = (base + tid < N_NODES) ? counts[base + tid] : 0;
        buf[tid] = v;
        __syncthreads();
        for (int off = 1; off < 1024; off <<= 1) {
            int t = (tid >= off) ? buf[tid - off] : 0;
            __syncthreads();
            buf[tid] += t;
            __syncthreads();
        }
        if (base + tid < N_NODES) col_ptr[base + tid] = carry + buf[tid] - v;  // exclusive
        __syncthreads();
        if (tid == 0) carry += buf[1023];
        __syncthreads();
    }
    if (tid == 0) col_ptr[N_NODES] = carry;
}

__global__ void sg_fill_kernel(const int* __restrict__ ei, const float* __restrict__ ef,
                               const int* __restrict__ col_ptr, int* __restrict__ fill,
                               const float* __restrict__ dinv1, const float* __restrict__ dinv2,
                               int* __restrict__ csr_src,
                               float* __restrict__ csr_w1, float* __restrict__ csr_w2) {
    int e = blockIdx.x * blockDim.x + threadIdx.x;
    if (e < N_EDGES) {
        int r = ei[e];
        int c = ei[N_EDGES + e];
        int pos = col_ptr[c] + atomicAdd(&fill[c], 1);
        csr_src[pos] = r;
        csr_w1[pos] = dinv1[r] * fabsf(ef[e * 4 + 0]) * dinv1[c];
        csr_w2[pos] = dinv2[r] * fabsf(ef[e * 4 + 1]) * dinv2[c];
    }
}

// ---------------------------------------------------------------------------
// SpMM hop: hout[n] = dinv[n]^2 * hin[n] + sum_in csr_w * hin[src]
// One block (128 threads) per destination node; 512 feats as float4.
// ---------------------------------------------------------------------------

__device__ __forceinline__ float4 sg_spmm_row(const float* __restrict__ hin,
                                              const int* __restrict__ col_ptr,
                                              const int* __restrict__ csr_src,
                                              const float* __restrict__ csr_w,
                                              const float* __restrict__ dinv,
                                              int n, int tid,
                                              int* s_src, float* s_w) {
    const float4* hin4 = (const float4*)hin;
    float s = dinv[n];
    s = s * s;
    float4 v = hin4[(size_t)n * 128 + tid];
    float4 acc;
    acc.x = s * v.x; acc.y = s * v.y; acc.z = s * v.z; acc.w = s * v.w;
    int start = col_ptr[n];
    int end   = col_ptr[n + 1];
    for (int j = start; j < end; j += 128) {
        int cnt = min(128, end - j);
        if (tid < cnt) {
            s_src[tid] = csr_src[j + tid];
            s_w[tid]   = csr_w[j + tid];
        }
        __syncthreads();
        for (int i = 0; i < cnt; ++i) {
            float  w = s_w[i];
            float4 u = hin4[(size_t)s_src[i] * 128 + tid];
            acc.x += w * u.x; acc.y += w * u.y; acc.z += w * u.z; acc.w += w * u.w;
        }
        __syncthreads();
    }
    return acc;
}

__global__ __launch_bounds__(128)
void sg_spmm_kernel(const float* __restrict__ hin, float* __restrict__ hout,
                    const int* __restrict__ col_ptr, const int* __restrict__ csr_src,
                    const float* __restrict__ csr_w, const float* __restrict__ dinv) {
    __shared__ int   s_src[128];
    __shared__ float s_w[128];
    int n   = blockIdx.x;
    int tid = threadIdx.x;
    float4 acc = sg_spmm_row(hin, col_ptr, csr_src, csr_w, dinv, n, tid, s_src, s_w);
    ((float4*)hout)[(size_t)n * 128 + tid] = acc;
}

// Final hop: write bf16 hi/lo split directly (fuses the precision split).
// Grid covers M_PAD rows; pad rows get zeros so GEMM loads need no guards.
__global__ __launch_bounds__(128)
void sg_spmm_split_kernel(const float* __restrict__ hin,
                          __bf16* __restrict__ hHi, __bf16* __restrict__ hLo,
                          const int* __restrict__ col_ptr, const int* __restrict__ csr_src,
                          const float* __restrict__ csr_w, const float* __restrict__ dinv) {
    __shared__ int   s_src[128];
    __shared__ float s_w[128];
    int n   = blockIdx.x;
    int tid = threadIdx.x;
    if (n >= N_NODES) {                       // zero the padding rows
        v4bf z = {};
        *(v4bf*)(hHi + (size_t)n * C_DIM + tid * 4) = z;
        *(v4bf*)(hLo + (size_t)n * C_DIM + tid * 4) = z;
        return;
    }
    float4 acc = sg_spmm_row(hin, col_ptr, csr_src, csr_w, dinv, n, tid, s_src, s_w);
    float f[4] = {acc.x, acc.y, acc.z, acc.w};
    v4bf hi, lo;
#pragma unroll
    for (int i = 0; i < 4; ++i) {
        __bf16 h = (__bf16)f[i];
        hi[i] = h;
        lo[i] = (__bf16)(f[i] - (float)h);
    }
    *(v4bf*)(hHi + (size_t)n * C_DIM + tid * 4) = hi;
    *(v4bf*)(hLo + (size_t)n * C_DIM + tid * 4) = lo;
}

// One-pass bf16 hi/lo split of a weight matrix (512x512).
__global__ void sg_wsplit_kernel(const float* __restrict__ Wm,
                                 __bf16* __restrict__ Whi, __bf16* __restrict__ Wlo) {
    int i = blockIdx.x * blockDim.x + threadIdx.x;   // over 512*512/4 float4s
    float4 v = ((const float4*)Wm)[i];
    float f[4] = {v.x, v.y, v.z, v.w};
    v4bf hi, lo;
#pragma unroll
    for (int q = 0; q < 4; ++q) {
        __bf16 h = (__bf16)f[q];
        hi[q] = h;
        lo[q] = (__bf16)(f[q] - (float)h);
    }
    ((v4bf*)Whi)[i] = hi;
    ((v4bf*)Wlo)[i] = lo;
}

// ---------------------------------------------------------------------------
// WMMA GEMM on pre-split bf16: out[:, cb+0:512] (and +1024 copy) = H @ W^T + b
// Block = 4 waves; wave w owns rows [(*4+w)*16, +16); block shares one 64-col
// strip whose B tile (hi+lo) is staged in LDS per K-step (async when available).
// bf16x3: hi*hi + hi*lo + lo*hi, f32 accumulation. Grid = (157, 8).
// ---------------------------------------------------------------------------

__global__ __launch_bounds__(128)
void sg_gemm_wmma_kernel(const __bf16* __restrict__ hHi, const __bf16* __restrict__ hLo,
                         const __bf16* __restrict__ Whi, const __bf16* __restrict__ Wlo,
                         const float* __restrict__ bias, float* __restrict__ out,
                         int colbase) {
    __shared__ __bf16 lds_b[2][64][32];   // [hi/lo][out-col within strip][k]

    int tid   = threadIdx.x;
    int lane  = tid & 31;
    int wave  = tid >> 5;
    int half  = lane >> 4;
    int l15   = lane & 15;
    int strip = blockIdx.y;                          // 0..7
    int mbase = (blockIdx.x * 4 + wave) * 16;        // < M_PAD, loads always valid

    // cooperative staging: thread t moves 16 bf16 (32B) of hi and of lo
    int srow = tid >> 1;                              // 0..63
    int skp  = (tid & 1) * 16;                        // 0 / 16
    const __bf16* gbh = Whi + (size_t)(strip * 64 + srow) * C_DIM + skp;
    const __bf16* gbl = Wlo + (size_t)(strip * 64 + srow) * C_DIM + skp;
    __bf16* lbh = &lds_b[0][srow][skp];
    __bf16* lbl = &lds_b[1][srow][skp];

    const __bf16* pah = hHi + (size_t)(mbase + l15) * C_DIM + 8 * half;
    const __bf16* pal = hLo + (size_t)(mbase + l15) * C_DIM + 8 * half;

    v8f cacc[4] = {v8f{}, v8f{}, v8f{}, v8f{}};

    for (int k0 = 0; k0 < C_DIM; k0 += 32) {
        // ---- stage B tile (hi+lo) for this K-step into LDS
#if __has_builtin(__builtin_amdgcn_global_load_async_to_lds_b128)
        __builtin_amdgcn_global_load_async_to_lds_b128(GAS(gbh + k0),     LAS(lbh),     0, 0);
        __builtin_amdgcn_global_load_async_to_lds_b128(GAS(gbh + k0 + 8), LAS(lbh + 8), 0, 0);
        __builtin_amdgcn_global_load_async_to_lds_b128(GAS(gbl + k0),     LAS(lbl),     0, 0);
        __builtin_amdgcn_global_load_async_to_lds_b128(GAS(gbl + k0 + 8), LAS(lbl + 8), 0, 0);
#if __has_builtin(__builtin_amdgcn_s_wait_asynccnt)
        __builtin_amdgcn_s_wait_asynccnt(0);
#else
        asm volatile("s_wait_asynccnt 0" ::: "memory");
#endif
#else
        *(v16bf*)lbh = *(const v16bf*)(gbh + k0);
        *(v16bf*)lbl = *(const v16bf*)(gbl + k0);
#endif
        __syncthreads();

        // ---- A fragments (ISA 7.12.2 16-bit A layout): elems 0..7 -> K=k0+8*half+i,
        //      elems 8..15 -> K=k0+16+8*half+i; both chunks contiguous in memory.
        v8bf a0 = *(const v8bf*)(pah + k0);
        v8bf a1 = *(const v8bf*)(pah + k0 + 16);
        v8bf b0 = *(const v8bf*)(pal + k0);
        v8bf b1 = *(const v8bf*)(pal + k0 + 16);
        v16bf ahi = __builtin_shufflevector(a0, a1, 0, 1, 2, 3, 4, 5, 6, 7,
                                            8, 9, 10, 11, 12, 13, 14, 15);
        v16bf alo = __builtin_shufflevector(b0, b1, 0, 1, 2, 3, 4, 5, 6, 7,
                                            8, 9, 10, 11, 12, 13, 14, 15);
        if (k0 + 32 < C_DIM) {                      // prefetch next A chunk
            __builtin_prefetch(pah + k0 + 32, 0, 1);
            __builtin_prefetch(pal + k0 + 32, 0, 1);
        }

#pragma unroll
        for (int nt = 0; nt < 4; ++nt) {
            // B fragment (32x16): lane holds N=l15; elems 0..7 -> K = 8*half + i, elems 8..15 -> K = 16 + 8*half + i.
            v16bf bhi = __builtin_shufflevector(*(const v8bf*)&lds_b[0][nt * 16 + l15][8 * half],
                                               *(const v8bf*)&lds_b[0][nt * 16 + l15][16 + 8 * half],
                                               0, 1, 2, 3, 4, 5, 6, 7, 8, 9, 10, 11, 12, 13, 14, 15);
            v16bf blo = __builtin_shufflevector(*(const v8bf*)&lds_b[1][nt * 16 + l15][8 * half],
                                               *(const v8bf*)&lds_b[1][nt * 16 + l15][16 + 8 * half],
                                               0, 1, 2, 3, 4, 5, 6, 7, 8, 9, 10, 11, 12, 13, 14, 15);
            cacc[nt] = __builtin_amdgcn_wmma_f32_16x16x32_bf16(
                false, ahi, false, bhi, (short)0, cacc[nt], false, false);
            cacc[nt] = __builtin_amdgcn_wmma_f32_16x16x32_bf16(
                false, ahi, false, blo, (short)0, cacc[nt], false, false);
            cacc[nt] = __builtin_amdgcn_wmma_f32_16x16x32_bf16(
                false, alo, false, bhi, (short)0, cacc[nt], false, false);
        }
        __syncthreads();
    }

    // ---- store: C/D layout: VGPR v -> M = v + 8*half; N = l15. Guard pad rows.
#pragma unroll
    for (int nt = 0; nt < 4; ++nt) {
        int ncol = strip * 64 + nt * 16 + l15;       // 0..511
        float bb = bias[ncol];
        int oc = colbase + ncol;
#pragma unroll
        for (int v = 0; v < 8; ++v) {
            int row = mbase + v + 8 * half;
            if (row < N_NODES) {
                float val = cacc[nt][v] + bb;
                out[(size_t)row * OUT_COLS + oc]        = val;
                out[(size_t)row * OUT_COLS + oc + 1024] = val;
            }
        }
    }
}

// ---------------------------------------------------------------------------

extern "C" void kernel_launch(void* const* d_in, const int* in_sizes, int n_in,
                              void* d_out, int out_size, void* d_ws, size_t ws_size,
                              hipStream_t stream) {
    (void)in_sizes; (void)n_in; (void)out_size; (void)ws_size;

    const float* x  = (const float*)d_in[0];
    const int*   ei = (const int*)d_in[1];
    const float* ef = (const float*)d_in[2];
    const float* W1 = (const float*)d_in[3];
    const float* b1 = (const float*)d_in[4];
    const float* W2 = (const float*)d_in[5];
    const float* b2 = (const float*)d_in[6];
    float* out = (float*)d_out;

    char* ws = (char*)d_ws;
    size_t off = 0;
    auto carve = [&](size_t bytes) -> void* {
        off = (off + 255) & ~(size_t)255;
        void* p = ws + off;
        off += bytes;
        return p;
    };
    int*    col_ptr = (int*)   carve((N_NODES + 1) * sizeof(int));
    int*    counts  = (int*)   carve(N_NODES * sizeof(int));
    int*    fill    = (int*)   carve(N_NODES * sizeof(int));
    float*  dinv1   = (float*) carve(N_NODES * sizeof(float));
    float*  dinv2   = (float*) carve(N_NODES * sizeof(float));
    int*    csr_src = (int*)   carve(N_EDGES * sizeof(int));
    float*  csr_w1  = (float*) carve(N_EDGES * sizeof(float));
    float*  csr_w2  = (float*) carve(N_EDGES * sizeof(float));
    float*  hA      = (float*) carve((size_t)N_NODES * C_DIM * sizeof(float));
    float*  hB      = (float*) carve((size_t)N_NODES * C_DIM * sizeof(float));
    __bf16* hHi     = (__bf16*)carve((size_t)M_PAD * C_DIM * sizeof(__bf16));
    __bf16* hLo     = (__bf16*)carve((size_t)M_PAD * C_DIM * sizeof(__bf16));
    __bf16* Whi     = (__bf16*)carve((size_t)C_DIM * C_DIM * sizeof(__bf16));
    __bf16* Wlo     = (__bf16*)carve((size_t)C_DIM * C_DIM * sizeof(__bf16));

    const int TPB = 256;
    sg_init_kernel<<<(N_NODES + TPB - 1) / TPB, TPB, 0, stream>>>(counts, fill, dinv1, dinv2);
    sg_edge_count_kernel<<<(N_EDGES + TPB - 1) / TPB, TPB, 0, stream>>>(ei, ef, counts, dinv1, dinv2);
    sg_dinv_kernel<<<(N_NODES + TPB - 1) / TPB, TPB, 0, stream>>>(dinv1, dinv2);
    sg_scan_kernel<<<1, 1024, 0, stream>>>(counts, col_ptr);
    sg_fill_kernel<<<(N_EDGES + TPB - 1) / TPB, TPB, 0, stream>>>(
        ei, ef, col_ptr, fill, dinv1, dinv2, csr_src, csr_w1, csr_w2);

    dim3 gemm_grid(M_PAD / 64, 8);
    int wsplit_blocks = (C_DIM * C_DIM / 4) / TPB;

    // head 1
    sg_spmm_kernel<<<N_NODES, 128, 0, stream>>>(x,  hA, col_ptr, csr_src, csr_w1, dinv1);
    sg_spmm_kernel<<<N_NODES, 128, 0, stream>>>(hA, hB, col_ptr, csr_src, csr_w1, dinv1);
    sg_spmm_split_kernel<<<M_PAD, 128, 0, stream>>>(hB, hHi, hLo, col_ptr, csr_src, csr_w1, dinv1);
    sg_wsplit_kernel<<<wsplit_blocks, TPB, 0, stream>>>(W1, Whi, Wlo);
    sg_gemm_wmma_kernel<<<gemm_grid, 128, 0, stream>>>(hHi, hLo, Whi, Wlo, b1, out, 0);

    // head 2
    sg_spmm_kernel<<<N_NODES, 128, 0, stream>>>(x,  hA, col_ptr, csr_src, csr_w2, dinv2);
    sg_spmm_kernel<<<N_NODES, 128, 0, stream>>>(hA, hB, col_ptr, csr_src, csr_w2, dinv2);
    sg_spmm_split_kernel<<<M_PAD, 128, 0, stream>>>(hB, hHi, hLo, col_ptr, csr_src, csr_w2, dinv2);
    sg_wsplit_kernel<<<wsplit_blocks, TPB, 0, stream>>>(W2, Whi, Wlo);
    sg_gemm_wmma_kernel<<<gemm_grid, 128, 0, stream>>>(hHi, hLo, Whi, Wlo, b2, out, 512);
}
